// EGNNLayer_661424963982
// MI455X (gfx1250) — matrix operand order repaired and checked by reading, not compiled, not run on hardware
//
#include <hip/hip_runtime.h>
#include <hip/hip_bf16.h>
#include <math.h>

#define Hdim 128
#define Cdim 3

typedef __attribute__((ext_vector_type(16))) __bf16 bf16x16;
typedef __attribute__((ext_vector_type(8)))  __bf16 bf16x8;
typedef __attribute__((ext_vector_type(8)))  float  f32x8;

// d_ws layout (bytes):
//   [0, N*H*4)        : msg_agg accumulator (f32, zeroed by init kernel)
//   [N*H*4, ...)      : packed bf16 transposed weights, element offsets below
#define OFF_WE1T 0        // [128][256]  <- We1[0:256][:]
#define OFF_WE2T 32768    // [128][128]
#define OFF_WC1T 49152    // [128][128]
#define OFF_WN1T 65536    // [128][256]
#define OFF_WN2T 98304    // [128][128]
#define PACK_TOTAL 114688

__device__ __forceinline__ float silu_f(float v) {
    return v * (1.0f / (1.0f + __expf(-v)));
}

// A-fragment (bf16 16x32, wave32): lane = M row (both halves), per-lane 16 K values:
// elements 0..15 -> K = koff+0..15, koff = (lane>=16)?16:0
__device__ __forceinline__ bf16x16 load_a_f32(const float* p0, const float* p1) {
    float4 c0 = *(const float4*)(p0);
    float4 c1 = *(const float4*)(p0 + 4);
    float4 c2 = *(const float4*)(p0 + 8);
    float4 c3 = *(const float4*)(p0 + 12);
    bf16x16 a;
    a[0]=(__bf16)c0.x;  a[1]=(__bf16)c0.y;  a[2]=(__bf16)c0.z;  a[3]=(__bf16)c0.w;
    a[4]=(__bf16)c1.x;  a[5]=(__bf16)c1.y;  a[6]=(__bf16)c1.z;  a[7]=(__bf16)c1.w;
    a[8]=(__bf16)c2.x;  a[9]=(__bf16)c2.y;  a[10]=(__bf16)c2.z; a[11]=(__bf16)c2.w;
    a[12]=(__bf16)c3.x; a[13]=(__bf16)c3.y; a[14]=(__bf16)c3.z; a[15]=(__bf16)c3.w;
    return a;
}

__device__ __forceinline__ bf16x16 load_a_lds(const __bf16* p0, const __bf16* p1) {
    bf16x8 lo = *(const bf16x8*)p0;
    bf16x8 hi = *(const bf16x8*)(p0 + 8);
    bf16x16 a;
    #pragma unroll
    for (int i = 0; i < 8; ++i) { a[i] = lo[i]; a[8+i] = hi[i]; }
    return a;
}

// ---------------- init: zero msg_agg, seed x_out = x ----------------
__global__ void egnn_init_kernel(const float* __restrict__ x, float* __restrict__ msg,
                                 float* __restrict__ xout, int nMsg, int nX) {
    int idx = blockIdx.x * blockDim.x + threadIdx.x;
    if (idx < nMsg) msg[idx] = 0.0f;
    if (idx < nX)   xout[idx] = x[idx];
}

// ---------------- pack transposed bf16 weights into ws ----------------
__global__ void egnn_pack_kernel(const float* __restrict__ We1, const float* __restrict__ We2,
                                 const float* __restrict__ Wc1, const float* __restrict__ Wn1,
                                 const float* __restrict__ Wn2, __bf16* __restrict__ pack) {
    int idx = blockIdx.x * blockDim.x + threadIdx.x;
    if (idx >= PACK_TOTAL) return;
    float v;
    if (idx < OFF_WE2T) {                       // We1T [128][256] <- We1[k][n]
        int j = idx - OFF_WE1T; int n = j >> 8, k = j & 255;
        v = We1[k * Hdim + n];
    } else if (idx < OFF_WC1T) {                // We2T [128][128]
        int j = idx - OFF_WE2T; int n = j >> 7, k = j & 127;
        v = We2[k * Hdim + n];
    } else if (idx < OFF_WN1T) {                // Wc1T [128][128]
        int j = idx - OFF_WC1T; int n = j >> 7, k = j & 127;
        v = Wc1[k * Hdim + n];
    } else if (idx < OFF_WN2T) {                // Wn1T [128][256]
        int j = idx - OFF_WN1T; int n = j >> 8, k = j & 255;
        v = Wn1[k * Hdim + n];
    } else {                                    // Wn2T [128][128]
        int j = idx - OFF_WN2T; int n = j >> 7, k = j & 127;
        v = Wn2[k * Hdim + n];
    }
    pack[idx] = (__bf16)v;
}

// ---------------- fused edge pipeline: one wave per 32-edge tile (M=32 blocking) ----
__launch_bounds__(32)
__global__ void egnn_edge_kernel(
    const float* __restrict__ h, const float* __restrict__ x,
    const int* __restrict__ src, const int* __restrict__ dst,
    const float* __restrict__ We1,   // f32 [257][128], for row 256 (dist_sq column)
    const float* __restrict__ be1, const float* __restrict__ be2,
    const float* __restrict__ bc1, const float* __restrict__ Wc2,
    const __bf16* __restrict__ pack,
    float* __restrict__ msg, float* __restrict__ xout)
{
    __shared__ __bf16 mtile[32][136];   // activation tile, bf16, padded rows
    __shared__ int   s_src[32];
    __shared__ int   s_dst[32];
    __shared__ float s_dsq[32];
    __shared__ float s_rel[32][3];

    const int lane   = threadIdx.x;
    const int base   = blockIdx.x * 32;
    const int mcol   = lane & 15;     // N column (B/C layout) and M row (A layout)
    const int hiHalf = lane >> 4;
    const int aKoff  = hiHalf * 16;   // A-fragment K sub-offset
    const int bKoff  = hiHalf * 16;   // B-fragment K sub-offset

    {   // per-edge geometry: all 32 lanes, one edge each
        int e = base + lane;
        int s = src[e], d = dst[e];
        s_src[lane] = s; s_dst[lane] = d;
        float dsq = 0.f;
        #pragma unroll
        for (int c = 0; c < Cdim; ++c) {
            float r = x[s * Cdim + c] - x[d * Cdim + c];
            s_rel[lane][c] = r;
            dsq += r * r;
        }
        s_dsq[lane] = dsq;
    }
    __syncthreads();

    // ---- Layer 1: [32 x 257] @ We1 = [32 x 128]; K=256 via WMMA, dist_sq rank-1 update
    const int sr0 = s_src[mcol],      dr0 = s_dst[mcol];       // tile 0 rows
    const int sr1 = s_src[16 + mcol], dr1 = s_dst[16 + mcol];  // tile 1 rows

    f32x8 acc0[8] = {};
    f32x8 acc1[8] = {};
    #pragma unroll
    for (int ks = 0; ks < 8; ++ks) {        // ks outer: 16 independent acc chains
        const float* rp0 = (ks < 4) ? (h + (size_t)sr0 * Hdim) : (h + (size_t)dr0 * Hdim);
        const float* rp1 = (ks < 4) ? (h + (size_t)sr1 * Hdim) : (h + (size_t)dr1 * Hdim);
        int kb = (ks & 3) * 32 + aKoff;
        bf16x16 a0 = load_a_f32(rp0 + kb, rp0 + kb + 16);
        bf16x16 a1 = load_a_f32(rp1 + kb, rp1 + kb + 16);
        const __bf16* wcol = pack + OFF_WE1T + mcol * 256 + ks * 32 + bKoff;
        bf16x16 bFs[8];                     // preload whole B ks-slab: one load clause
        #pragma unroll
        for (int nt = 0; nt < 8; ++nt)
            bFs[nt] = *(const bf16x16*)(wcol + nt * (16 * 256));
        #pragma unroll
        for (int nt = 0; nt < 8; ++nt) {
            acc0[nt] = __builtin_amdgcn_wmma_f32_16x16x32_bf16(
                false, a0, false, bFs[nt], (short)0, acc0[nt], false, false);
            acc1[nt] = __builtin_amdgcn_wmma_f32_16x16x32_bf16(
                false, a1, false, bFs[nt], (short)0, acc1[nt], false, false);
        }
    }

    #pragma unroll
    for (int nt = 0; nt < 8; ++nt) {
        const int n = nt * 16 + mcol;
        float w256 = We1[256 * Hdim + n];
        float b1   = be1[n];
        #pragma unroll
        for (int r = 0; r < 8; ++r) {
            int M = r + 8 * hiHalf;
            mtile[M][n]      = (__bf16)silu_f(acc0[nt][r] + s_dsq[M] * w256 + b1);
            mtile[16 + M][n] = (__bf16)silu_f(acc1[nt][r] + s_dsq[16 + M] * w256 + b1);
        }
    }
    __syncthreads();

    // ---- Layer 2: m1 @ We2 (K=128) -> m_ij (SiLU) ; scatter-add into msg_agg
    #pragma unroll
    for (int nt = 0; nt < 8; ++nt) { acc0[nt] = (f32x8){}; acc1[nt] = (f32x8){}; }
    #pragma unroll
    for (int ks = 0; ks < 4; ++ks) {
        int kb = ks * 32 + aKoff;
        bf16x16 a0 = load_a_lds(&mtile[mcol][kb],      &mtile[mcol][kb + 16]);
        bf16x16 a1 = load_a_lds(&mtile[16 + mcol][kb], &mtile[16 + mcol][kb + 16]);
        const __bf16* wcol = pack + OFF_WE2T + mcol * 128 + ks * 32 + bKoff;
        bf16x16 bFs[8];
        #pragma unroll
        for (int nt = 0; nt < 8; ++nt)
            bFs[nt] = *(const bf16x16*)(wcol + nt * (16 * 128));
        #pragma unroll
        for (int nt = 0; nt < 8; ++nt) {
            acc0[nt] = __builtin_amdgcn_wmma_f32_16x16x32_bf16(
                false, a0, false, bFs[nt], (short)0, acc0[nt], false, false);
            acc1[nt] = __builtin_amdgcn_wmma_f32_16x16x32_bf16(
                false, a1, false, bFs[nt], (short)0, acc1[nt], false, false);
        }
    }
    __syncthreads();   // all LDS reads of layer-1 tile done before overwrite

    #pragma unroll
    for (int nt = 0; nt < 8; ++nt) {
        const int n = nt * 16 + mcol;
        float b2 = be2[n];
        #pragma unroll
        for (int r = 0; r < 8; ++r) {
            int M = r + 8 * hiHalf;
            float v0 = silu_f(acc0[nt][r] + b2);                    // m_ij[M][n]
            float v1 = silu_f(acc1[nt][r] + b2);                    // m_ij[16+M][n]
            atomicAdd(&msg[(size_t)s_dst[M] * Hdim + n], v0);       // segment_sum(m_ij)
            atomicAdd(&msg[(size_t)s_dst[16 + M] * Hdim + n], v1);
            mtile[M][n]      = (__bf16)v0;
            mtile[16 + M][n] = (__bf16)v1;
        }
    }
    __syncthreads();

    // ---- Layer 3: coord hidden = SiLU(m_ij @ Wc1 + bc1)
    #pragma unroll
    for (int nt = 0; nt < 8; ++nt) { acc0[nt] = (f32x8){}; acc1[nt] = (f32x8){}; }
    #pragma unroll
    for (int ks = 0; ks < 4; ++ks) {
        int kb = ks * 32 + aKoff;
        bf16x16 a0 = load_a_lds(&mtile[mcol][kb],      &mtile[mcol][kb + 16]);
        bf16x16 a1 = load_a_lds(&mtile[16 + mcol][kb], &mtile[16 + mcol][kb + 16]);
        const __bf16* wcol = pack + OFF_WC1T + mcol * 128 + ks * 32 + bKoff;
        bf16x16 bFs[8];
        #pragma unroll
        for (int nt = 0; nt < 8; ++nt)
            bFs[nt] = *(const bf16x16*)(wcol + nt * (16 * 128));
        #pragma unroll
        for (int nt = 0; nt < 8; ++nt) {
            acc0[nt] = __builtin_amdgcn_wmma_f32_16x16x32_bf16(
                false, a0, false, bFs[nt], (short)0, acc0[nt], false, false);
            acc1[nt] = __builtin_amdgcn_wmma_f32_16x16x32_bf16(
                false, a1, false, bFs[nt], (short)0, acc1[nt], false, false);
        }
    }
    __syncthreads();

    #pragma unroll
    for (int nt = 0; nt < 8; ++nt) {
        const int n = nt * 16 + mcol;
        float b3 = bc1[n];
        #pragma unroll
        for (int r = 0; r < 8; ++r) {
            int M = r + 8 * hiHalf;
            mtile[M][n]      = (__bf16)silu_f(acc0[nt][r] + b3);
            mtile[16 + M][n] = (__bf16)silu_f(acc1[nt][r] + b3);
        }
    }
    __syncthreads();

    // ---- coord_weight = hidden @ Wc2 (128 -> 1); scatter rel_pos * w into x_out
    {
        float wsum = 0.f;
        #pragma unroll 8
        for (int k = 0; k < Hdim; ++k) wsum += (float)mtile[lane][k] * Wc2[k];
        int d = s_dst[lane];
        #pragma unroll
        for (int c = 0; c < Cdim; ++c)
            atomicAdd(&xout[d * Cdim + c], s_rel[lane][c] * wsum);
    }
}

// ---------------- node update: one wave per 32-node tile ----------------
__launch_bounds__(32)
__global__ void egnn_node_kernel(
    const float* __restrict__ h, const float* __restrict__ msg,
    const float* __restrict__ bn1, const float* __restrict__ bn2,
    const __bf16* __restrict__ pack, float* __restrict__ hout, int Nn)
{
    __shared__ __bf16 mtile[32][136];

    const int lane   = threadIdx.x;
    const int base   = blockIdx.x * 32;
    const int mcol   = lane & 15;
    const int hiHalf = lane >> 4;
    const int aKoff  = hiHalf * 16;
    const int bKoff  = hiHalf * 16;
    const int row0   = base + mcol;
    const int row1r  = base + 16 + mcol;
    const int row1   = (row1r < Nn) ? row1r : (Nn - 1);   // clamp loads; guard stores

    // node_input = [h | msg_agg], K = 256
    f32x8 acc0[8] = {};
    f32x8 acc1[8] = {};
    #pragma unroll
    for (int ks = 0; ks < 8; ++ks) {
        const float* rp0 = (ks < 4) ? (h + (size_t)row0 * Hdim) : (msg + (size_t)row0 * Hdim);
        const float* rp1 = (ks < 4) ? (h + (size_t)row1 * Hdim) : (msg + (size_t)row1 * Hdim);
        int kb = (ks & 3) * 32 + aKoff;
        bf16x16 a0 = load_a_f32(rp0 + kb, rp0 + kb + 16);
        bf16x16 a1 = load_a_f32(rp1 + kb, rp1 + kb + 16);
        const __bf16* wcol = pack + OFF_WN1T + mcol * 256 + ks * 32 + bKoff;
        bf16x16 bFs[8];
        #pragma unroll
        for (int nt = 0; nt < 8; ++nt)
            bFs[nt] = *(const bf16x16*)(wcol + nt * (16 * 256));
        #pragma unroll
        for (int nt = 0; nt < 8; ++nt) {
            acc0[nt] = __builtin_amdgcn_wmma_f32_16x16x32_bf16(
                false, a0, false, bFs[nt], (short)0, acc0[nt], false, false);
            acc1[nt] = __builtin_amdgcn_wmma_f32_16x16x32_bf16(
                false, a1, false, bFs[nt], (short)0, acc1[nt], false, false);
        }
    }

    #pragma unroll
    for (int nt = 0; nt < 8; ++nt) {
        const int n = nt * 16 + mcol;
        float b1 = bn1[n];
        #pragma unroll
        for (int r = 0; r < 8; ++r) {
            int M = r + 8 * hiHalf;
            mtile[M][n]      = (__bf16)silu_f(acc0[nt][r] + b1);
            mtile[16 + M][n] = (__bf16)silu_f(acc1[nt][r] + b1);
        }
    }
    __syncthreads();

    #pragma unroll
    for (int nt = 0; nt < 8; ++nt) { acc0[nt] = (f32x8){}; acc1[nt] = (f32x8){}; }
    #pragma unroll
    for (int ks = 0; ks < 4; ++ks) {
        int kb = ks * 32 + aKoff;
        bf16x16 a0 = load_a_lds(&mtile[mcol][kb],      &mtile[mcol][kb + 16]);
        bf16x16 a1 = load_a_lds(&mtile[16 + mcol][kb], &mtile[16 + mcol][kb + 16]);
        const __bf16* wcol = pack + OFF_WN2T + mcol * 128 + ks * 32 + bKoff;
        bf16x16 bFs[8];
        #pragma unroll
        for (int nt = 0; nt < 8; ++nt)
            bFs[nt] = *(const bf16x16*)(wcol + nt * (16 * 128));
        #pragma unroll
        for (int nt = 0; nt < 8; ++nt) {
            acc0[nt] = __builtin_amdgcn_wmma_f32_16x16x32_bf16(
                false, a0, false, bFs[nt], (short)0, acc0[nt], false, false);
            acc1[nt] = __builtin_amdgcn_wmma_f32_16x16x32_bf16(
                false, a1, false, bFs[nt], (short)0, acc1[nt], false, false);
        }
    }

    // h_out = h + (hidden @ Wn2 + bn2)
    #pragma unroll
    for (int nt = 0; nt < 8; ++nt) {
        const int n = nt * 16 + mcol;
        float b2 = bn2[n];
        #pragma unroll
        for (int r = 0; r < 8; ++r) {
            int M = r + 8 * hiHalf;
            size_t gi0 = (size_t)(base + M) * Hdim + n;
            hout[gi0] = acc0[nt][r] + b2 + h[gi0];
            int g1 = base + 16 + M;
            if (g1 < Nn) {
                size_t gi1 = (size_t)g1 * Hdim + n;
                hout[gi1] = acc1[nt][r] + b2 + h[gi1];
            }
        }
    }
}

extern "C" void kernel_launch(void* const* d_in, const int* in_sizes, int n_in,
                              void* d_out, int out_size, void* d_ws, size_t ws_size,
                              hipStream_t stream) {
    const float* h   = (const float*)d_in[0];
    const float* x   = (const float*)d_in[1];
    const int*   ei  = (const int*)d_in[2];
    const float* We1 = (const float*)d_in[3];
    const float* be1 = (const float*)d_in[4];
    const float* We2 = (const float*)d_in[5];
    const float* be2 = (const float*)d_in[6];
    const float* Wc1 = (const float*)d_in[7];
    const float* bc1 = (const float*)d_in[8];
    const float* Wc2 = (const float*)d_in[9];
    const float* Wn1 = (const float*)d_in[10];
    const float* bn1 = (const float*)d_in[11];
    const float* Wn2 = (const float*)d_in[12];
    const float* bn2 = (const float*)d_in[13];

    const int N = in_sizes[0] / Hdim;
    const int E = in_sizes[2] / 2;
    const int* src = ei;
    const int* dst = ei + E;

    float* hout = (float*)d_out;
    float* xout = hout + (size_t)N * Hdim;

    float*  msg  = (float*)d_ws;
    __bf16* pack = reinterpret_cast<__bf16*>((char*)d_ws + (size_t)N * Hdim * sizeof(float));

    const int nMsg = N * Hdim;
    const int nX   = N * Cdim;
    egnn_init_kernel<<<(nMsg + 255) / 256, 256, 0, stream>>>(x, msg, xout, nMsg, nX);
    egnn_pack_kernel<<<(PACK_TOTAL + 255) / 256, 256, 0, stream>>>(We1, We2, Wc1, Wn1, Wn2, pack);
    egnn_edge_kernel<<<E / 32, 32, 0, stream>>>(h, x, src, dst, We1, be1, be2, bc1, Wc2,
                                                pack, msg, xout);
    egnn_node_kernel<<<(N + 31) / 32, 32, 0, stream>>>(h, msg, bn1, bn2, pack, hout, N);
}
